// MultiHeadAttention_83141976916905
// MI455X (gfx1250) — matrix operand order repaired and checked by reading, not compiled, not run on hardware
//
#include <hip/hip_runtime.h>
#include <hip/hip_bf16.h>
#include <math.h>

// ---- problem sizes -------------------------------------------------------
#define D_MODEL 1024
#define NUM_HEADS 16
#define D_K 64
#define BATCH 4
#define SEQ 2048
#define MROWS (BATCH * SEQ)      // 8192
#define N_QKV (3 * D_MODEL)      // 3072

// ---- CDNA5 WMMA types ----------------------------------------------------
typedef __attribute__((ext_vector_type(16))) _Float16 v16h;
typedef __attribute__((ext_vector_type(8)))  float    v8f;
typedef __attribute__((ext_vector_type(4)))  unsigned int v4u;
typedef __attribute__((ext_vector_type(8)))  int      v8i;
typedef __attribute__((ext_vector_type(4)))  int      v4i;

// 5-arg TDM builtin lane (ROCm 7.2); therock toolchain (6-arg) ships the TDM header.
#if __has_builtin(__builtin_amdgcn_tensor_load_to_lds) && !__has_include(<hip/amd_detail/amd_gfx1250_TDM.h>)
#define USE_TDM 1
#pragma message("CDNA5 probe: TDM 5-arg path ENABLED")
#else
#define USE_TDM 0
#pragma message("CDNA5 probe: TDM path DISABLED (builtin missing or 6-arg header present)")
#endif

union Frag { v16h h; v4u u[2]; };

// A-matrix fragment (16x32 f16, M x K): lane (l&15)=row M; two 16B loads at
// halfword offsets hi*8 and 16+hi*8 (ISA 16-bit A layout).
__device__ __forceinline__ v16h load_frag_A(const _Float16* p, int hi) {
  Frag f;
  f.u[0] = *(const v4u*)(p + hi * 8);
  f.u[1] = *(const v4u*)(p + hi * 8 + 16);
  return f.h;
}

// B-matrix fragment (32x16 f16, K x N) from a K-contiguous row (memory holds
// B^T rows): lane (l&15)=col N; two 16B loads at halfword offsets hi*8, +16.
__device__ __forceinline__ v16h load_frag_B(const _Float16* p, int hi) {
  Frag f;
  f.u[0] = *(const v4u*)(p + hi * 8);
  f.u[1] = *(const v4u*)(p + hi * 8 + 16);
  return f.h;
}

__device__ __forceinline__ v8f wmma_f16(v16h a, v16h b, v8f c) {
  return __builtin_amdgcn_wmma_f32_16x16x32_f16(
      false, a, false, b, (short)0, c, false, false);
}

// Cross-lane reductions within 16-lane groups (wave32) via ds_swizzle
// group-of-32 mode: and=0x1f, or=0, xor=M -> offset = 31 | (M<<10).
template <int M>
__device__ __forceinline__ float swz_max(float x) {
  int y = __builtin_amdgcn_ds_swizzle(__builtin_bit_cast(int, x), 31 | (M << 10));
  return fmaxf(x, __builtin_bit_cast(float, y));
}
template <int M>
__device__ __forceinline__ float swz_add(float x) {
  int y = __builtin_amdgcn_ds_swizzle(__builtin_bit_cast(int, x), 31 | (M << 10));
  return x + __builtin_bit_cast(float, y);
}

// CDNA5 async copy: per-lane 16B global -> LDS, tracked with ASYNCcnt.
__device__ __forceinline__ void async_b128(_Float16* ldst, const _Float16* gsrc) {
  unsigned lds_addr = (unsigned)(size_t)ldst;   // low 32 bits = LDS byte offset
  asm volatile("global_load_async_to_lds_b128 %0, %1, off"
               :: "v"(lds_addr), "v"(gsrc)
               : "memory");
}
__device__ __forceinline__ void wait_asynccnt0() {
  asm volatile("s_wait_asynccnt 0x0" ::: "memory");
}

#if USE_TDM
// Tensor Data Mover: 2D tile load (f16 elements) per ISA ch.8 D# layout.
// Tensor = [tensor_d1 lines][tensor_d0 elems], line stride stride0 (elems).
// Tile   = [tile_d1 lines][tile_d0 elems], packed contiguously into LDS.
__device__ __forceinline__ void tdm_load_2d(unsigned lds_addr,
                                            const _Float16* gaddr,
                                            unsigned tensor_d0, unsigned tensor_d1,
                                            unsigned stride0,
                                            unsigned tile_d0, unsigned tile_d1) {
  unsigned long long ga = (unsigned long long)(size_t)gaddr;
  v4u g0;
  g0[0] = 1u;                                            // count=1, user D#
  g0[1] = lds_addr;                                      // lds_addr [63:32]
  g0[2] = (unsigned)(ga & 0xffffffffu);                  // global_addr lo
  g0[3] = (unsigned)((ga >> 32) & 0x1ffffffu) | (2u << 30);  // addr hi | type=2
  v8i g1 = {};
  g1[0] = (int)(1u << 16);                               // data_size=1 (2B)
  g1[1] = (int)((tensor_d0 & 0xffffu) << 16);            // tensor_dim0[15:0]
  g1[2] = (int)(((tensor_d0 >> 16) & 0xffffu) |          // tensor_dim0[31:16]
                ((tensor_d1 & 0xffffu) << 16));          // tensor_dim1[15:0]
  g1[3] = (int)(((tensor_d1 >> 16) & 0xffffu) |          // tensor_dim1[31:16]
                ((tile_d0 & 0xffffu) << 16));            // tile_dim0
  g1[4] = (int)(tile_d1 & 0xffffu);                      // tile_dim1 (tile_dim2=0)
  g1[5] = (int)stride0;                                  // tensor_dim0_stride lo32
  v4i g2 = {};
  v4i g3 = {};
  __builtin_amdgcn_tensor_load_to_lds(g0, g1, g2, g3, 0);
}
#endif

// ---- kernel 1: f32 -> f16 conversion ------------------------------------
__global__ void cvt_f32_f16(const float* __restrict__ src,
                            _Float16* __restrict__ dst, int n) {
  int i = blockIdx.x * blockDim.x + threadIdx.x;
  int stride = gridDim.x * blockDim.x;
  for (; i < n; i += stride) dst[i] = (_Float16)src[i];
}

// ---- kernel 2: fused QKV projection GEMM --------------------------------
// QKV[m][n] = sum_k X[m][k] * Wqkv[n][k] + b[n].  One wave = 16x64 tile.
// Ping-pong fragment buffers (even/odd K-step) -> pipelined loads with no
// register-rotation copies.
__global__ void __launch_bounds__(256) qkv_gemm(
    const _Float16* __restrict__ X, const _Float16* __restrict__ W,
    const float* __restrict__ bias, _Float16* __restrict__ Q,
    _Float16* __restrict__ K, _Float16* __restrict__ Vt) {
  const int tid = threadIdx.x;
  const int lane = tid & 31, hi = lane >> 4, lo16 = lane & 15;
  const int waveG = blockIdx.x * (blockDim.x >> 5) + (tid >> 5);
  const int NGRPS = N_QKV / 64;                 // 48
  const int mTile = waveG / NGRPS;
  const int nGrp = waveG % NGRPS;
  const int m0 = mTile * 16, n0 = nGrp * 64;

  v8f acc[4] = {};
  const _Float16* aRow = X + (size_t)(m0 + lo16) * D_MODEL;
  const _Float16* bRow = W + (size_t)(n0 + lo16) * D_MODEL;

  v16h a0 = load_frag_A(aRow, hi);
  v16h b0[4], a1, b1[4];
#pragma unroll
  for (int nt = 0; nt < 4; ++nt)
    b0[nt] = load_frag_B(bRow + (size_t)nt * 16 * D_MODEL, hi);

  for (int kk2 = 0; kk2 < 15; ++kk2) {
    const int k1 = 2 * kk2 + 1, k2 = 2 * kk2 + 2;
    a1 = load_frag_A(aRow + k1 * 32, hi);
#pragma unroll
    for (int nt = 0; nt < 4; ++nt)
      b1[nt] = load_frag_B(bRow + (size_t)nt * 16 * D_MODEL + k1 * 32, hi);
#pragma unroll
    for (int nt = 0; nt < 4; ++nt) acc[nt] = wmma_f16(a0, b0[nt], acc[nt]);
    a0 = load_frag_A(aRow + k2 * 32, hi);
#pragma unroll
    for (int nt = 0; nt < 4; ++nt)
      b0[nt] = load_frag_B(bRow + (size_t)nt * 16 * D_MODEL + k2 * 32, hi);
#pragma unroll
    for (int nt = 0; nt < 4; ++nt) acc[nt] = wmma_f16(a1, b1[nt], acc[nt]);
  }
  // tail: k = 30 (in a0/b0) and k = 31
  a1 = load_frag_A(aRow + 31 * 32, hi);
#pragma unroll
  for (int nt = 0; nt < 4; ++nt)
    b1[nt] = load_frag_B(bRow + (size_t)nt * 16 * D_MODEL + 31 * 32, hi);
#pragma unroll
  for (int nt = 0; nt < 4; ++nt) acc[nt] = wmma_f16(a0, b0[nt], acc[nt]);
#pragma unroll
  for (int nt = 0; nt < 4; ++nt) acc[nt] = wmma_f16(a1, b1[nt], acc[nt]);

#pragma unroll
  for (int nt = 0; nt < 4; ++nt) {
    const int nG = n0 + nt * 16 + lo16;
    const float bv = bias[nG];
    const int h = nG / (3 * D_K);
    const int c = nG % (3 * D_K);
#pragma unroll
    for (int r = 0; r < 8; ++r) {
      const int mG = m0 + r + 8 * hi;
      const int b_ = mG >> 11;                  // / SEQ
      const int s = mG & (SEQ - 1);
      const int bh = b_ * NUM_HEADS + h;
      const float v = acc[nt][r] + bv;
      if (c < D_K)
        Q[((size_t)bh * SEQ + s) * D_K + c] = (_Float16)v;
      else if (c < 2 * D_K)
        K[((size_t)bh * SEQ + s) * D_K + (c - D_K)] = (_Float16)v;
      else
        Vt[((size_t)bh * D_K + (c - 2 * D_K)) * SEQ + s] = (_Float16)v;
    }
  }
}

// ---- kernel 3: flash attention ------------------------------------------
// grid = (BATCH*NUM_HEADS, SEQ/128); block = 256 (8 waves, wave32).
// Each wave owns a 16-row q tile; 8 waves share one K/Vt LDS chunk (32 kv).
// K chunk staged with async-to-LDS; Vt chunk (strided 2D tile) via the TDM.
__global__ void __launch_bounds__(256) attn_kernel(
    const _Float16* __restrict__ Q, const _Float16* __restrict__ K,
    const _Float16* __restrict__ Vt, const float* __restrict__ mask,
    _Float16* __restrict__ O) {
  __shared__ __align__(16) _Float16 ldsK[32 * 64];   // [kv][d]
  __shared__ __align__(16) _Float16 ldsV[64 * 32];   // [d][kv]
  __shared__ __align__(16) _Float16 ldsP[8 * 16 * 32];

  const int tid = threadIdx.x;
  const int lane = tid & 31, hi = lane >> 4, lo16 = lane & 15;
  const int wave = tid >> 5;
  const int bh = blockIdx.x;
  const int qBase = blockIdx.y * 128 + wave * 16;

  v16h qa0, qa1;
  {
    const _Float16* qRow = Q + ((size_t)bh * SEQ + qBase + lo16) * D_K;
    qa0 = load_frag_A(qRow, hi);
    qa1 = load_frag_A(qRow + 32, hi);
  }

  v8f o0 = {}, o1 = {}, o2 = {}, o3 = {};
  float mrow[8], lrow[8];
#pragma unroll
  for (int r = 0; r < 8; ++r) { mrow[r] = -__builtin_inff(); lrow[r] = 0.f; }

  const _Float16* Kbh = K + (size_t)bh * SEQ * D_K;
  const _Float16* Vbh = Vt + (size_t)bh * D_K * SEQ;
  _Float16* Pw = ldsP + wave * (16 * 32);

  for (int kv0 = 0; kv0 < SEQ; kv0 += 32) {
    // ---- stage K chunk [32x64] (contiguous 4KB) via async-to-LDS ---------
    {
      const int f = tid * 8;                    // 256 threads * 8 halfwords
      async_b128(ldsK + f, Kbh + (size_t)kv0 * D_K + f);
    }
    // ---- stage Vt chunk [64 rows x 32 cols], row stride SEQ --------------
#if USE_TDM
    if (wave == 0) {
      tdm_load_2d((unsigned)(size_t)&ldsV[0], Vbh + kv0,
                  /*tensor_d0=*/SEQ, /*tensor_d1=*/D_K,
                  /*stride0=*/SEQ, /*tile_d0=*/32, /*tile_d1=*/D_K);
      __builtin_amdgcn_s_wait_tensorcnt(0);
    }
#else
    {
      const int f = tid * 8;
      const int vd = f >> 5, vc = f & 31;
      async_b128(ldsV + f, Vbh + (size_t)vd * SEQ + kv0 + vc);
    }
#endif
    if (kv0 + 32 < SEQ) {
      __builtin_prefetch(Kbh + (size_t)(kv0 + 32 + (tid >> 3)) * D_K, 0, 1);
      __builtin_prefetch(Vbh + (size_t)(tid >> 2) * SEQ + kv0 + 32, 0, 1);
    }
    wait_asynccnt0();
    __syncthreads();

    // ---- scores S = Q K^T (two 16x16 col tiles, accumulated over d=64) ---
    v8f sc[2];
#pragma unroll
    for (int t = 0; t < 2; ++t) {
      const _Float16* kRow = ldsK + (t * 16 + lo16) * 64;
      v8f z = {};
      z = wmma_f16(qa0, load_frag_B(kRow, hi), z);
      sc[t] = wmma_f16(qa1, load_frag_B(kRow + 32, hi), z);
    }

    // ---- online softmax (C-layout: lane=col lo16, rows r + 8*hi) ---------
    float p0a[8], p1a[8], alpha[8];
#pragma unroll
    for (int r = 0; r < 8; ++r) {
      const int qRow = qBase + r + 8 * hi;
      const float s0 = sc[0][r] * 0.125f + mask[(size_t)qRow * SEQ + kv0 + lo16];
      const float s1 = sc[1][r] * 0.125f + mask[(size_t)qRow * SEQ + kv0 + 16 + lo16];
      float mnew = fmaxf(s0, s1);
      mnew = swz_max<1>(mnew); mnew = swz_max<2>(mnew);
      mnew = swz_max<4>(mnew); mnew = swz_max<8>(mnew);
      mnew = fmaxf(mnew, mrow[r]);
      alpha[r] = __expf(mrow[r] - mnew);
      mrow[r] = mnew;
      const float p0 = __expf(s0 - mnew);
      const float p1 = __expf(s1 - mnew);
      p0a[r] = p0; p1a[r] = p1;
      float rs = p0 + p1;
      rs = swz_add<1>(rs); rs = swz_add<2>(rs);
      rs = swz_add<4>(rs); rs = swz_add<8>(rs);
      lrow[r] = lrow[r] * alpha[r] + rs;
    }
#pragma unroll
    for (int r = 0; r < 8; ++r) {
      o0[r] *= alpha[r]; o1[r] *= alpha[r];
      o2[r] *= alpha[r]; o3[r] *= alpha[r];
      Pw[(r + 8 * hi) * 32 + lo16]      = (_Float16)p0a[r];
      Pw[(r + 8 * hi) * 32 + 16 + lo16] = (_Float16)p1a[r];
    }
    // ---- O += P V --------------------------------------------------------
    v16h pa = load_frag_A(Pw + lo16 * 32, hi);
    o0 = wmma_f16(pa, load_frag_B(ldsV + (0 * 16 + lo16) * 32, hi), o0);
    o1 = wmma_f16(pa, load_frag_B(ldsV + (1 * 16 + lo16) * 32, hi), o1);
    o2 = wmma_f16(pa, load_frag_B(ldsV + (2 * 16 + lo16) * 32, hi), o2);
    o3 = wmma_f16(pa, load_frag_B(ldsV + (3 * 16 + lo16) * 32, hi), o3);
    __syncthreads();
  }

  // ---- normalize and write O (f16, [B,S,H*D_K]) --------------------------
  const int b_ = bh >> 4, h = bh & 15;
#pragma unroll
  for (int r = 0; r < 8; ++r) {
    const float inv = 1.0f / lrow[r];
    const int s = qBase + r + 8 * hi;
    _Float16* orow = O + ((size_t)b_ * SEQ + s) * D_MODEL + h * D_K + lo16;
    orow[0]  = (_Float16)(o0[r] * inv);
    orow[16] = (_Float16)(o1[r] * inv);
    orow[32] = (_Float16)(o2[r] * inv);
    orow[48] = (_Float16)(o3[r] * inv);
  }
}

// ---- kernel 4: output projection GEMM (f32 out) -------------------------
__global__ void __launch_bounds__(256) out_gemm(
    const _Float16* __restrict__ A, const _Float16* __restrict__ W,
    const float* __restrict__ bias, float* __restrict__ out) {
  const int tid = threadIdx.x;
  const int lane = tid & 31, hi = lane >> 4, lo16 = lane & 15;
  const int waveG = blockIdx.x * (blockDim.x >> 5) + (tid >> 5);
  const int NGRPS = D_MODEL / 64;               // 16
  const int mTile = waveG / NGRPS;
  const int nGrp = waveG % NGRPS;
  const int m0 = mTile * 16, n0 = nGrp * 64;

  v8f acc[4] = {};
  const _Float16* aRow = A + (size_t)(m0 + lo16) * D_MODEL;
  const _Float16* bRow = W + (size_t)(n0 + lo16) * D_MODEL;

  v16h a0 = load_frag_A(aRow, hi);
  v16h b0[4], a1, b1[4];
#pragma unroll
  for (int nt = 0; nt < 4; ++nt)
    b0[nt] = load_frag_B(bRow + (size_t)nt * 16 * D_MODEL, hi);

  for (int kk2 = 0; kk2 < 15; ++kk2) {
    const int k1 = 2 * kk2 + 1, k2 = 2 * kk2 + 2;
    a1 = load_frag_A(aRow + k1 * 32, hi);
#pragma unroll
    for (int nt = 0; nt < 4; ++nt)
      b1[nt] = load_frag_B(bRow + (size_t)nt * 16 * D_MODEL + k1 * 32, hi);
#pragma unroll
    for (int nt = 0; nt < 4; ++nt) acc[nt] = wmma_f16(a0, b0[nt], acc[nt]);
    a0 = load_frag_A(aRow + k2 * 32, hi);
#pragma unroll
    for (int nt = 0; nt < 4; ++nt)
      b0[nt] = load_frag_B(bRow + (size_t)nt * 16 * D_MODEL + k2 * 32, hi);
#pragma unroll
    for (int nt = 0; nt < 4; ++nt) acc[nt] = wmma_f16(a1, b1[nt], acc[nt]);
  }
  a1 = load_frag_A(aRow + 31 * 32, hi);
#pragma unroll
  for (int nt = 0; nt < 4; ++nt)
    b1[nt] = load_frag_B(bRow + (size_t)nt * 16 * D_MODEL + 31 * 32, hi);
#pragma unroll
  for (int nt = 0; nt < 4; ++nt) acc[nt] = wmma_f16(a0, b0[nt], acc[nt]);
#pragma unroll
  for (int nt = 0; nt < 4; ++nt) acc[nt] = wmma_f16(a1, b1[nt], acc[nt]);

#pragma unroll
  for (int nt = 0; nt < 4; ++nt) {
    const int nG = n0 + nt * 16 + lo16;
    const float bv = bias[nG];
#pragma unroll
    for (int r = 0; r < 8; ++r) {
      const int mG = m0 + r + 8 * hi;
      out[(size_t)mG * D_MODEL + nG] = acc[nt][r] + bv;
    }
  }
}

// ---- host launch ---------------------------------------------------------
extern "C" void kernel_launch(void* const* d_in, const int* in_sizes, int n_in,
                              void* d_out, int out_size, void* d_ws,
                              size_t ws_size, hipStream_t stream) {
  const float* x      = (const float*)d_in[0];   // [B,S,D]
  const float* mask   = (const float*)d_in[1];   // [1,1,S,S]
  const float* W_qkv  = (const float*)d_in[2];   // [3D,D]
  const float* b_qkv  = (const float*)d_in[3];   // [3D]
  const float* W_o    = (const float*)d_in[4];   // [D,D]
  const float* b_o    = (const float*)d_in[5];   // [D]
  float* out = (float*)d_out;

  // workspace layout (bytes); total = 88 MiB
  char* ws = (char*)d_ws;
  _Float16* Xh    = (_Float16*)(ws + 0);                         // 8192*1024
  _Float16* Wqkvh = (_Float16*)(ws + (size_t)16 * 1024 * 1024);  // 3072*1024
  _Float16* Woh   = (_Float16*)(ws + (size_t)22 * 1024 * 1024);  // 1024*1024
  _Float16* Qg    = (_Float16*)(ws + (size_t)24 * 1024 * 1024);  // 64*2048*64
  _Float16* Kg    = (_Float16*)(ws + (size_t)40 * 1024 * 1024);
  _Float16* Vtg   = (_Float16*)(ws + (size_t)56 * 1024 * 1024);
  _Float16* Oh    = (_Float16*)(ws + (size_t)72 * 1024 * 1024);

  cvt_f32_f16<<<2048, 256, 0, stream>>>(x, Xh, MROWS * D_MODEL);
  cvt_f32_f16<<<2048, 256, 0, stream>>>(W_qkv, Wqkvh, N_QKV * D_MODEL);
  cvt_f32_f16<<<1024, 256, 0, stream>>>(W_o, Woh, D_MODEL * D_MODEL);

  qkv_gemm<<<3072, 256, 0, stream>>>(Xh, Wqkvh, b_qkv, Qg, Kg, Vtg);

  attn_kernel<<<dim3(BATCH * NUM_HEADS, SEQ / 128), 256, 0, stream>>>(
      Qg, Kg, Vtg, mask, Oh);

  out_gemm<<<1024, 256, 0, stream>>>(Oh, Woh, b_o, out);
}
